// GNN_v2_5927054868944
// MI455X (gfx1250) — compile-verified, matrix operand order checked by reading, not run on hardware
//
#include <hip/hip_runtime.h>
#include <hip/hip_bf16.h>

#define NEG_SLOPE 0.2f
#define NUM_GRAPHS 256
#define ENC_NEG_INF 0x007fffffu

typedef _Float16 v16h __attribute__((ext_vector_type(16)));
typedef _Float16 v8h  __attribute__((ext_vector_type(8)));
typedef float    v8f  __attribute__((ext_vector_type(8)));

// ---- order-preserving float<->uint map for atomic max on signed floats ----
__device__ __forceinline__ unsigned enc_ord(float f) {
    unsigned u = __float_as_uint(f);
    return (u & 0x80000000u) ? ~u : (u | 0x80000000u);
}
__device__ __forceinline__ float dec_ord(unsigned u) {
    unsigned b = (u & 0x80000000u) ? (u & 0x7fffffffu) : ~u;
    return __uint_as_float(b);
}
__device__ __forceinline__ float elu_f(float v) { return v > 0.f ? v : (__expf(v) - 1.f); }

// ------------------------------------------------------------------ init ----
__global__ void init_nodes_k(unsigned* emax_u, float* denom, float* deg,
                             float* x1, float* x2, int N) {
    int t = blockIdx.x * blockDim.x + threadIdx.x;
    if (t >= N * 32) return;
    int n = t >> 5, k = t & 31;
    x2[t] = 0.f;
    if (k < 16) x1[n * 16 + k] = 0.f;
    if (k == 0) { emax_u[n] = ENC_NEG_INF; denom[n] = 0.f; deg[n] = 0.f; }
}

__global__ void init_pool_k(unsigned* pool_max_u, float* pool_sum, float* counts) {
    int t = blockIdx.x * blockDim.x + threadIdx.x;
    if (t >= NUM_GRAPHS * 32) return;
    pool_max_u[t] = ENC_NEG_INF;
    pool_sum[t] = 0.f;
    if ((t & 31) == 0) counts[t >> 5] = 0.f;
}

// ------------------------------------------------------- GAT node features --
// h = x @ gat_W  (input dim 1 -> h[n][k] = x[n]*W[k]);  a_src/a_dst = h . att
__global__ void gat_node_k(const float* __restrict__ x, const float* __restrict__ gat_W,
                           const float* __restrict__ att_s, const float* __restrict__ att_d,
                           float* __restrict__ h, float* __restrict__ a_s, float* __restrict__ a_d,
                           int N) {
    int n = blockIdx.x * blockDim.x + threadIdx.x;
    if (n >= N) return;
    float xv = x[n];
    float as = 0.f, ad = 0.f;
#pragma unroll
    for (int k = 0; k < 16; ++k) {
        float hv = xv * gat_W[k];
        h[n * 16 + k] = hv;
        as += hv * att_s[k];
        ad += hv * att_d[k];
    }
    a_s[n] = as;
    a_d[n] = ad;
}

// ------------------------------------------------------------- edge phases --
// Edge t < E comes from edge_index; t in [E, E+N) is the self loop (t-E, t-E).
__global__ void edge_max_k(const int* __restrict__ ei, int E, int N,
                           const float* __restrict__ a_s, const float* __restrict__ a_d,
                           unsigned* __restrict__ emax_u) {
    int t = blockIdx.x * blockDim.x + threadIdx.x;
    if (t >= E + N) return;
    int s = (t < E) ? ei[t]     : (t - E);
    int d = (t < E) ? ei[E + t] : (t - E);
    float e = a_s[s] + a_d[d];
    e = e > 0.f ? e : NEG_SLOPE * e;
    atomicMax(emax_u + d, enc_ord(e));
}

__global__ void edge_sum_k(const int* __restrict__ ei, int E, int N,
                           const float* __restrict__ a_s, const float* __restrict__ a_d,
                           const unsigned* __restrict__ emax_u,
                           float* __restrict__ ex, float* __restrict__ denom,
                           float* __restrict__ deg) {
    int t = blockIdx.x * blockDim.x + threadIdx.x;
    if (t >= E + N) return;
    int s = (t < E) ? ei[t]     : (t - E);
    int d = (t < E) ? ei[E + t] : (t - E);
    float e = a_s[s] + a_d[d];
    e = e > 0.f ? e : NEG_SLOPE * e;
    float v = __expf(e - dec_ord(emax_u[d]));
    ex[t] = v;
    atomicAdd(denom + d, v);
    atomicAdd(deg + d, 1.f);
}

__global__ void edge_gat_agg_k(const int* __restrict__ ei, int E, int N,
                               const float* __restrict__ ex, const float* __restrict__ denom,
                               const float* __restrict__ h, float* __restrict__ x1) {
    int t = blockIdx.x * blockDim.x + threadIdx.x;
    if (t >= E + N) return;
    int s = (t < E) ? ei[t]     : (t - E);
    int d = (t < E) ? ei[E + t] : (t - E);
    float alpha = ex[t] / (denom[d] + 1e-16f);
    const float4* hp = reinterpret_cast<const float4*>(h + (size_t)s * 16);
    float4 h0 = hp[0], h1 = hp[1], h2v = hp[2], h3 = hp[3];
    float hv[16] = {h0.x, h0.y, h0.z, h0.w, h1.x, h1.y, h1.z, h1.w,
                    h2v.x, h2v.y, h2v.z, h2v.w, h3.x, h3.y, h3.z, h3.w};
    float* xd = x1 + (size_t)d * 16;
#pragma unroll
    for (int k = 0; k < 16; ++k) atomicAdd(xd + k, hv[k] * alpha);
}

// x1 = elu(x1 + gat_b)  (store f16 copy for WMMA); dinv = deg^-1/2
__global__ void node_fin1_k(const float* __restrict__ x1, const float* __restrict__ gat_b,
                            const float* __restrict__ deg, _Float16* __restrict__ x1h,
                            float* __restrict__ dinv, int N) {
    int t = blockIdx.x * blockDim.x + threadIdx.x;
    if (t >= N * 16) return;
    int n = t >> 4, k = t & 15;
    float v = elu_f(x1[t] + gat_b[k]);
    x1h[t] = (_Float16)v;
    if (k == 0) {
        float dv = deg[n];
        dinv[n] = dv > 0.f ? rsqrtf(dv) : 0.f;
    }
}

// ------------------------------------------- GCN dense GEMM via CDNA5 WMMA --
// h2[N,32] = x1h[N,16] @ gcn_W[16,32], one wave per 16-row tile,
// v_wmma_f32_16x16x32_f16 with K=16 real (K=16..31 zero padded).
__global__ void gcn_gemm_wmma_k(const _Float16* __restrict__ x1h,
                                const float* __restrict__ gcn_W,
                                float* __restrict__ h2, int N, int nTiles) {
    int wid  = threadIdx.x >> 5;                     // wave in block (8 waves)
    int lane = threadIdx.x & 31;
    int tile = blockIdx.x * 8 + wid;
    if (tile >= nTiles) return;
    int row0 = tile * 16;
    int r  = lane & 15;
    int hi = lane >> 4;                              // half-wave select
    int row = row0 + r;
    bool full = (row0 + 16 <= N);                    // wave-uniform fast path

    // A image (16-bit A 16x32, ISA 7.12.2): lanes 0-15 elems 0..7 = K0..7,
    // lanes 16-31 elems 0..7 = K8..15; elems 8..15 (K16..31) stay zero.
    v16h A = {};
    if (row < N) {
        v8h av = *reinterpret_cast<const v8h*>(x1h + (size_t)row * 16 + hi * 8);
#pragma unroll
        for (int i = 0; i < 8; ++i) A[i] = av[i];
    }

#pragma unroll
    for (int nt = 0; nt < 2; ++nt) {
        int col = nt * 16 + r;
        v16h B = {};                                 // B image mirrors A along K
#pragma unroll
        for (int i = 0; i < 8; ++i) B[i] = (_Float16)gcn_W[(hi * 8 + i) * 32 + col];
        v8f C = {};
        C = __builtin_amdgcn_wmma_f32_16x16x32_f16(
                /*neg_a=*/false, A, /*neg_b=*/false, B,
                /*c_mod=*/(short)0, C, /*reuse_a=*/false, /*reuse_b=*/false);
        // C layout: lane<16 -> M = i; lane>=16 -> M = i+8; N = lane%16
        float* outp = h2 + ((size_t)(row0 + hi * 8)) * 32 + col;
        if (full) {                                  // branch-free store path
#pragma unroll
            for (int i = 0; i < 8; ++i) outp[(size_t)i * 32] = C[i];
        } else {                                     // ragged tail (unused for N%16==0)
#pragma unroll
            for (int i = 0; i < 8; ++i)
                if (row0 + hi * 8 + i < N) outp[(size_t)i * 32] = C[i];
        }
    }
}

__global__ void edge_gcn_agg_k(const int* __restrict__ ei, int E, int N,
                               const float* __restrict__ dinv,
                               const float* __restrict__ h2, float* __restrict__ x2) {
    int t = blockIdx.x * blockDim.x + threadIdx.x;
    if (t >= E + N) return;
    int s = (t < E) ? ei[t]     : (t - E);
    int d = (t < E) ? ei[E + t] : (t - E);
    float norm = dinv[s] * dinv[d];
    const float4* hp = reinterpret_cast<const float4*>(h2 + (size_t)s * 32);
    float* xd = x2 + (size_t)d * 32;
#pragma unroll
    for (int q = 0; q < 8; ++q) {
        float4 v = hp[q];
        atomicAdd(xd + q * 4 + 0, v.x * norm);
        atomicAdd(xd + q * 4 + 1, v.y * norm);
        atomicAdd(xd + q * 4 + 2, v.z * norm);
        atomicAdd(xd + q * 4 + 3, v.w * norm);
    }
}

// x2 = elu(x2 + gcn_b); pool max/sum/count per graph
__global__ void node_fin2_pool_k(const float* __restrict__ x2, const float* __restrict__ gcn_b,
                                 const int* __restrict__ batch,
                                 unsigned* __restrict__ pool_max_u, float* __restrict__ pool_sum,
                                 float* __restrict__ counts, int N) {
    int t = blockIdx.x * blockDim.x + threadIdx.x;
    if (t >= N * 32) return;
    int n = t >> 5, k = t & 31;
    float v = elu_f(x2[t] + gcn_b[k]);
    int g = batch[n];
    atomicMax(pool_max_u + g * 32 + k, enc_ord(v));
    atomicAdd(pool_sum + g * 32 + k, v);
    if (k == 0) atomicAdd(counts + g, 1.f);
}

// feats = [max | mean | add] (96) ; out = feats @ lin_W + lin_b
__global__ void final_k(const unsigned* __restrict__ pool_max_u, const float* __restrict__ pool_sum,
                        const float* __restrict__ counts, const float* __restrict__ lin_W,
                        const float* __restrict__ lin_b, float* __restrict__ out) {
    int g = blockIdx.x;
    int t = threadIdx.x;
    __shared__ float feats[96];
    if (t < 96) {
        int k = t % 32;
        float v;
        if (t < 32)       v = dec_ord(pool_max_u[g * 32 + k]);
        else if (t < 64)  v = pool_sum[g * 32 + k] / fmaxf(counts[g], 1.f);
        else              v = pool_sum[g * 32 + k];
        feats[t] = v;
    }
    __syncthreads();
    if (t < 2) {
        float acc = lin_b[t];
#pragma unroll 8
        for (int j = 0; j < 96; ++j) acc += feats[j] * lin_W[j * 2 + t];
        out[g * 2 + t] = acc;
    }
}

// ----------------------------------------------------------------- launch ---
extern "C" void kernel_launch(void* const* d_in, const int* in_sizes, int n_in,
                              void* d_out, int out_size, void* d_ws, size_t ws_size,
                              hipStream_t stream) {
    const float* x       = (const float*)d_in[0];
    const float* gat_W   = (const float*)d_in[1];
    const float* att_src = (const float*)d_in[2];
    const float* att_dst = (const float*)d_in[3];
    const float* gat_b   = (const float*)d_in[4];
    const float* gcn_W   = (const float*)d_in[5];
    const float* gcn_b   = (const float*)d_in[6];
    const float* lin_W   = (const float*)d_in[7];
    const float* lin_b   = (const float*)d_in[8];
    const int*   ei      = (const int*)d_in[9];
    const int*   batch   = (const int*)d_in[10];
    float* out = (float*)d_out;

    const int N  = in_sizes[0];       // x is [N,1]
    const int E  = in_sizes[9] / 2;   // edge_index is [2,E]
    const int ET = E + N;             // edges + self loops

    // carve workspace (256B aligned regions)
    char* p = (char*)d_ws;
    auto carve = [&](size_t bytes) -> char* {
        char* r = p;
        p += (bytes + 255) & ~(size_t)255;
        return r;
    };
    float*     h        = (float*)carve((size_t)N * 16 * 4);
    float*     a_s      = (float*)carve((size_t)N * 4);
    float*     a_d      = (float*)carve((size_t)N * 4);
    unsigned*  emax_u   = (unsigned*)carve((size_t)N * 4);
    float*     denom    = (float*)carve((size_t)N * 4);
    float*     deg      = (float*)carve((size_t)N * 4);
    float*     dinv     = (float*)carve((size_t)N * 4);
    float*     ex       = (float*)carve((size_t)ET * 4);
    float*     x1       = (float*)carve((size_t)N * 16 * 4);
    _Float16*  x1h      = (_Float16*)carve((size_t)N * 16 * 2);
    float*     h2       = (float*)carve((size_t)N * 32 * 4);
    float*     x2       = (float*)carve((size_t)N * 32 * 4);
    unsigned*  pmax     = (unsigned*)carve((size_t)NUM_GRAPHS * 32 * 4);
    float*     psum     = (float*)carve((size_t)NUM_GRAPHS * 32 * 4);
    float*     cnts     = (float*)carve((size_t)NUM_GRAPHS * 4);
    (void)ws_size; (void)n_in; (void)out_size;

    const int TB = 256;
    auto nb = [&](long long n) { return (int)((n + TB - 1) / TB); };

    init_nodes_k<<<nb((long long)N * 32), TB, 0, stream>>>(emax_u, denom, deg, x1, x2, N);
    init_pool_k<<<nb(NUM_GRAPHS * 32), TB, 0, stream>>>(pmax, psum, cnts);

    gat_node_k<<<nb(N), TB, 0, stream>>>(x, gat_W, att_src, att_dst, h, a_s, a_d, N);

    edge_max_k<<<nb(ET), TB, 0, stream>>>(ei, E, N, a_s, a_d, emax_u);
    edge_sum_k<<<nb(ET), TB, 0, stream>>>(ei, E, N, a_s, a_d, emax_u, ex, denom, deg);
    edge_gat_agg_k<<<nb(ET), TB, 0, stream>>>(ei, E, N, ex, denom, h, x1);

    node_fin1_k<<<nb((long long)N * 16), TB, 0, stream>>>(x1, gat_b, deg, x1h, dinv, N);

    int nTiles = (N + 15) / 16;
    gcn_gemm_wmma_k<<<(nTiles + 7) / 8, 256, 0, stream>>>(x1h, gcn_W, h2, N, nTiles);

    edge_gcn_agg_k<<<nb(ET), TB, 0, stream>>>(ei, E, N, dinv, h2, x2);

    node_fin2_pool_k<<<nb((long long)N * 32), TB, 0, stream>>>(x2, gcn_b, batch, pmax, psum, cnts, N);

    final_k<<<NUM_GRAPHS, 128, 0, stream>>>(pmax, psum, cnts, lin_W, lin_b, out);
}
